// Net1_2_88081189306910
// MI455X (gfx1250) — matrix operand order repaired and checked by reading, not compiled, not run on hardware
//
#include <hip/hip_runtime.h>

typedef __attribute__((ext_vector_type(4)))  _Float16 h4;
typedef __attribute__((ext_vector_type(8)))  _Float16 h8;
typedef __attribute__((ext_vector_type(16))) _Float16 h16;
typedef __attribute__((ext_vector_type(8)))  float    f8;

// ---------------- degree count ----------------
__global__ void count_kernel(const long long* __restrict__ dst, float* __restrict__ cnt,
                             long long E) {
    long long e = (long long)blockIdx.x * blockDim.x + threadIdx.x;
    if (e >= E) return;
    atomicAdd(&cnt[dst[e]], 1.0f);
}

// ---------------- scatter-add (f32 source) ----------------
__global__ void scatter_add_f32(const long long* __restrict__ src,
                                const long long* __restrict__ dst,
                                const float* __restrict__ feat,
                                float* __restrict__ agg,
                                long long E, int d) {
    long long tid = (long long)blockIdx.x * blockDim.x + threadIdx.x;
    int g = d >> 2;                       // float4 groups per row
    long long total = E * (long long)g;
    if (tid >= total) return;
    long long e = tid / g;
    int f = (int)(tid % g) << 2;
    long long s = src[e], t = dst[e];
    float4 v = *(const float4*)(feat + s * (long long)d + f);
    float* o = agg + t * (long long)d + f;
    atomicAdd(o + 0, v.x); atomicAdd(o + 1, v.y);
    atomicAdd(o + 2, v.z); atomicAdd(o + 3, v.w);
}

// ---------------- scatter-add (f16 source) ----------------
__global__ void scatter_add_f16(const long long* __restrict__ src,
                                const long long* __restrict__ dst,
                                const _Float16* __restrict__ feat,
                                float* __restrict__ agg,
                                long long E, int d) {
    long long tid = (long long)blockIdx.x * blockDim.x + threadIdx.x;
    int g = d >> 2;
    long long total = E * (long long)g;
    if (tid >= total) return;
    long long e = tid / g;
    int f = (int)(tid % g) << 2;
    long long s = src[e], t = dst[e];
    h4 v = *(const h4*)(feat + s * (long long)d + f);
    float* o = agg + t * (long long)d + f;
    atomicAdd(o + 0, (float)v.x); atomicAdd(o + 1, (float)v.y);
    atomicAdd(o + 2, (float)v.z); atomicAdd(o + 3, (float)v.w);
}

// ---------------- finalize: A = [agg/cnt | h]  (f16, row-major, K = 2*din) ----------------
__global__ void finalize_kernel(const float* __restrict__ agg,
                                const float* __restrict__ cnt,
                                const float* __restrict__ h32,      // either this
                                const _Float16* __restrict__ h16p,  // or this is non-null
                                _Float16* __restrict__ A,
                                long long N, int din) {
    long long tid = (long long)blockIdx.x * blockDim.x + threadIdx.x;
    long long total = N * (long long)din;
    if (tid >= total) return;
    long long n = tid / din;
    int f = (int)(tid % din);
    float inv = 1.0f / fmaxf(cnt[n], 1.0f);
    long long K = 2 * (long long)din;
    A[n * K + f] = (_Float16)(agg[tid] * inv);
    float hv = h32 ? h32[tid] : (float)h16p[tid];
    A[n * K + din + f] = (_Float16)hv;
}

// ---------------- pack [Wl; Wr] -> WMMA B-fragment layout (f16) ----------------
// Bp index = ((tn*kBlocks + kb)*32 + lane)*16 + i
// element i of lane l maps to K = kb*32 + (l<16 ? 0 : 8) + (i<8 ? i : 8+i), Ncol = tn*16 + (l&15)
__global__ void pack_weights(const float* __restrict__ Wl, const float* __restrict__ Wr,
                             _Float16* __restrict__ Bp, int din, int dout) {
    int K = 2 * din;
    int kBlocks = K >> 5;
    long long total = (long long)dout * K;
    long long tid = (long long)blockIdx.x * blockDim.x + threadIdx.x;
    if (tid >= total) return;
    int i  = (int)(tid & 15);
    int l  = (int)((tid >> 4) & 31);
    int kb = (int)((tid >> 9) % kBlocks);
    int tn = (int)(tid / ((long long)kBlocks << 9));
    int kk = kb * 32 + ((l < 16) ? 0 : 8) + ((i < 8) ? i : 8 + i);
    int n  = tn * 16 + (l & 15);
    float v = (kk < din) ? Wl[(long long)kk * dout + n]
                         : Wr[(long long)(kk - din) * dout + n];
    Bp[tid] = (_Float16)v;
}

// ---------------- fused dual-GEMM + bias (+ReLU), f16 in / f32 acc / f16 out ----------------
// one wave32 per 16x16 output tile; v_wmma_f32_16x16x32_f16 over K blocks of 32
__global__ __launch_bounds__(256) void gemm_wmma_f16(
    const _Float16* __restrict__ A,   // [N, K] row-major f16
    const _Float16* __restrict__ Bp,  // packed fragments
    const float* __restrict__ bias,   // [dout]
    _Float16* __restrict__ H,         // [N, dout] f16 out
    int nTilesM, int nTilesN, int K, int dout, int doRelu) {
    int wave = threadIdx.x >> 5;
    int lane = threadIdx.x & 31;
    int tile = blockIdx.x * 8 + wave;
    int total = nTilesM * nTilesN;
    if (tile >= total) return;                 // wave-uniform: EXEC all-1s inside
    int tm = tile / nTilesN;
    int tn = tile - tm * nTilesN;

    int row = tm * 16 + (lane & 15);
    const _Float16* ar = A + (long long)row * K;
    int aoff = (lane < 16) ? 0 : 8;            // ISA 16-bit A layout
    int kBlocks = K >> 5;
    const _Float16* bp = Bp + ((long long)tn * kBlocks << 9) + ((long long)lane << 4);

    f8 c = {};
    for (int kb = 0; kb < kBlocks; ++kb) {
        const _Float16* ak = ar + (kb << 5) + aoff;
        h8 alo = *(const h8*)(ak);             // K = aoff .. aoff+7
        h8 ahi = *(const h8*)(ak + 16);        // K = aoff+16 .. aoff+23
        h16 a = __builtin_shufflevector(alo, ahi,
                                        0, 1, 2, 3, 4, 5, 6, 7,
                                        8, 9, 10, 11, 12, 13, 14, 15);
        h16 b = *(const h16*)(bp + ((long long)kb << 9));
        c = __builtin_amdgcn_wmma_f32_16x16x32_f16(false, a, false, b,
                                                   (short)0, c, false, false);
    }

    int col = tn * 16 + (lane & 15);
    float bv = bias[col];
    int rbase = tm * 16 + ((lane < 16) ? 0 : 8);
#pragma unroll
    for (int r = 0; r < 8; ++r) {
        float v = c[r] + bv;
        if (doRelu) v = v > 0.0f ? v : 0.0f;
        H[(long long)(rbase + r) * dout + col] = (_Float16)v;
    }
}

// ---------------- layer 3: dout=1 GEMV, wave-per-node, f32 ----------------
__global__ void layer3_kernel(const float* __restrict__ agg,
                              const float* __restrict__ cnt,
                              const _Float16* __restrict__ h2,
                              const float* __restrict__ Wl,
                              const float* __restrict__ bl,
                              const float* __restrict__ Wr,
                              float* __restrict__ out, long long N) {
    int wave = threadIdx.x >> 5;
    int lane = threadIdx.x & 31;
    long long n = (long long)blockIdx.x * 8 + wave;
    if (n >= N) return;
    float inv = 1.0f / fmaxf(cnt[n], 1.0f);
    float s = 0.0f;
    for (int f = lane; f < 128; f += 32) {
        s += agg[n * 128 + f] * inv * Wl[f];
        s += (float)h2[n * 128 + f] * Wr[f];
    }
    for (int m = 16; m; m >>= 1) s += __shfl_xor(s, m, 32);
    if (lane == 0) out[n] = s + bl[0];
}

extern "C" void kernel_launch(void* const* d_in, const int* in_sizes, int n_in,
                              void* d_out, int out_size, void* d_ws, size_t ws_size,
                              hipStream_t stream) {
    const float*     x   = (const float*)d_in[0];
    const long long* ei  = (const long long*)d_in[1];  // int64 edge_index [2, E]
    const float* Wl1 = (const float*)d_in[2];
    const float* bl1 = (const float*)d_in[3];
    const float* Wr1 = (const float*)d_in[4];
    const float* Wl2 = (const float*)d_in[5];
    const float* bl2 = (const float*)d_in[6];
    const float* Wr2 = (const float*)d_in[7];
    const float* Wl3 = (const float*)d_in[8];
    const float* bl3 = (const float*)d_in[9];
    const float* Wr3 = (const float*)d_in[10];
    float* out = (float*)d_out;

    const int d1 = 256, d2 = 128;
    long long N = (long long)in_sizes[0] / d1;
    long long E = (long long)in_sizes[1] / 2;
    const long long* srcIdx = ei;
    const long long* dstIdx = ei + E;

    // ---- workspace carve-up (256B aligned) ----
    char* ws = (char*)d_ws;
    auto al = [](size_t v) { return (v + 255) & ~(size_t)255; };
    size_t off = 0;
    float*    cnt = (float*)(ws + off);    off = al(off + (size_t)N * 4);
    float*    agg = (float*)(ws + off);    off = al(off + (size_t)N * d1 * 4);
    _Float16* Abuf = (_Float16*)(ws + off); off = al(off + (size_t)N * 2 * d1 * 2);
    _Float16* h1  = (_Float16*)(ws + off); off = al(off + (size_t)N * d1 * 2);
    _Float16* h2  = (_Float16*)(ws + off); off = al(off + (size_t)N * d2 * 2);
    _Float16* Bp1 = (_Float16*)(ws + off); off = al(off + (size_t)2 * d1 * d1 * 2);
    _Float16* Bp2 = (_Float16*)(ws + off); off = al(off + (size_t)2 * d1 * d2 * 2);
    (void)ws_size; (void)n_in; (void)out_size;

    auto blocksFor = [](long long total) { return (unsigned)((total + 255) / 256); };

    // degree counts (shared by all layers)
    hipMemsetAsync(cnt, 0, (size_t)N * 4, stream);
    count_kernel<<<blocksFor(E), 256, 0, stream>>>(dstIdx, cnt, E);

    // pack weights
    pack_weights<<<blocksFor((long long)2 * d1 * d1), 256, 0, stream>>>(Wl1, Wr1, Bp1, d1, d1);
    pack_weights<<<blocksFor((long long)2 * d1 * d2), 256, 0, stream>>>(Wl2, Wr2, Bp2, d1, d2);

    // ---- layer 1 ----
    hipMemsetAsync(agg, 0, (size_t)N * d1 * 4, stream);
    scatter_add_f32<<<blocksFor(E * (d1 / 4)), 256, 0, stream>>>(srcIdx, dstIdx, x, agg, E, d1);
    finalize_kernel<<<blocksFor(N * d1), 256, 0, stream>>>(agg, cnt, x, nullptr, Abuf, N, d1);
    {
        int tM = (int)(N / 16), tN = d1 / 16;
        gemm_wmma_f16<<<(unsigned)((tM * tN + 7) / 8), 256, 0, stream>>>(
            Abuf, Bp1, bl1, h1, tM, tN, 2 * d1, d1, 1);
    }

    // ---- layer 2 ----
    hipMemsetAsync(agg, 0, (size_t)N * d1 * 4, stream);
    scatter_add_f16<<<blocksFor(E * (d1 / 4)), 256, 0, stream>>>(srcIdx, dstIdx, h1, agg, E, d1);
    finalize_kernel<<<blocksFor(N * d1), 256, 0, stream>>>(agg, cnt, nullptr, h1, Abuf, N, d1);
    {
        int tM = (int)(N / 16), tN = d2 / 16;
        gemm_wmma_f16<<<(unsigned)((tM * tN + 7) / 8), 256, 0, stream>>>(
            Abuf, Bp2, bl2, h2, tM, tN, 2 * d1, d2, 1);
    }

    // ---- layer 3 (dout = 1) ----
    hipMemsetAsync(agg, 0, (size_t)N * d2 * 4, stream);
    scatter_add_f16<<<blocksFor(E * (d2 / 4)), 256, 0, stream>>>(srcIdx, dstIdx, h2, agg, E, d2);
    layer3_kernel<<<(unsigned)((N + 7) / 8), 256, 0, stream>>>(agg, cnt, h2, Wl3, bl3, Wr3, out, N);
}
